// CstcovModel_9053791060565
// MI455X (gfx1250) — compile-verified, matrix operand order checked by reading, not run on hardware
//
#include <hip/hip_runtime.h>
#include <hip/hip_bf16.h>
#include <math.h>

// ---------------------------------------------------------------------------
// CDNA5 / gfx1250 implementation of the CConv fluid network.
// wave32. WMMA f32<-f16 16x16x32 for both per-query GEMM stages, operand-major
// LDS layouts so every fragment is two 16B ds_load_b128, TDM async tensor load
// for the per-query interpolation tables, f16 pre-transposed filter bank in L2.
// ---------------------------------------------------------------------------

typedef __attribute__((ext_vector_type(16))) _Float16 v16h;
typedef __attribute__((ext_vector_type(8)))  _Float16 v8h;
typedef __attribute__((ext_vector_type(8)))  float    v8f;
typedef __attribute__((ext_vector_type(4)))  unsigned u32x4;
typedef __attribute__((ext_vector_type(8)))  int      i32x8;
typedef __attribute__((ext_vector_type(4)))  int      i32x4;

#define PI_F   3.14159265358979323846f
#define EPS_F  1e-8f
#define BIG_F  1e30f
#define DT_F   0.1f
#define RADIUS 3.0f      // EXTENT*0.5 = 40*6*0.025*0.5
#define R2_F   9.0f

__device__ __forceinline__ float sgnf(float v) {
    return (v > 0.f) ? 1.f : ((v < 0.f) ? -1.f : 0.f);
}

// ---------------------------------------------------------------------------
// TDM: 1-D contiguous DMA (global -> LDS), 4-byte elements.
// D# per CDNA5 ISA: group0 = {flags, lds_addr, gaddr},
// group1 = {data_size, tensor_dim0, tile_dim0}. Groups 2/3 zero (<=2D tensor).
// Verified to lower to `tensor_load_to_lds` + `s_wait_tensorcnt` in round 2.
// ---------------------------------------------------------------------------
#if defined(__HIP_DEVICE_COMPILE__) && __has_builtin(__builtin_amdgcn_tensor_load_to_lds)
#define HAVE_TDM 1
__device__ __forceinline__ void tdm_load_1d(const void* gptr, unsigned ldsOff,
                                            unsigned nElems) {
    unsigned long long ga = (unsigned long long)gptr;
    u32x4 g0 = {};
    g0[0] = 1u;                                   // count=1 (valid user D#)
    g0[1] = ldsOff;                               // lds_addr (bytes)
    g0[2] = (unsigned)(ga & 0xffffffffu);         // global_addr[31:0]
    g0[3] = (unsigned)((ga >> 32) & 0x1ffffffu)   // global_addr[56:32]
          | (2u << 30);                           // type=2 ("image")
    i32x8 g1 = {};
    g1[0] = (int)(2u << 16);                      // data_size=4B; mask/flags=0
    g1[1] = (int)((nElems & 0xffffu) << 16);      // tensor_dim0[15:0] @ bits 63:48
    g1[2] = (int)((nElems >> 16) & 0xffffu);      // tensor_dim0[31:16] @ bits 79:64
    g1[3] = (int)((nElems & 0xffffu) << 16);      // tile_dim0 @ bits 127:112
    i32x4 z4 = {};
#if __clang_major__ <= 22
    __builtin_amdgcn_tensor_load_to_lds(g0, g1, z4, z4, 0);
#else
    i32x8 z8 = {};
    __builtin_amdgcn_tensor_load_to_lds(g0, g1, z4, z4, z8, 0);
#endif
}
#endif

// ---------------------------------------------------------------------------
// 1. time integration + fluid feature assembly: feats = [v1, other, v0_enc]
// ---------------------------------------------------------------------------
__global__ void integrate_kernel(const float* __restrict__ p0,
                                 const float* __restrict__ v0,
                                 const float* __restrict__ a,
                                 const float* __restrict__ other,
                                 const float* __restrict__ v0enc,
                                 float* __restrict__ p1,
                                 float* __restrict__ v1,
                                 float* __restrict__ ff,   // (N,3,3)
                                 int N) {
    int n = blockIdx.x * blockDim.x + threadIdx.x;
    if (n >= N) return;
    for (int d = 0; d < 3; ++d) {
        float vv1 = v0[n*3+d] + DT_F * a[n*3+d];
        float pp1 = p0[n*3+d] + DT_F * (v0[n*3+d] + vv1) * 0.5f;
        v1[n*3+d] = vv1;
        p1[n*3+d] = pp1;
        ff[(n*3+0)*3+d] = vv1;
        ff[(n*3+1)*3+d] = other[n*3+d];
        ff[(n*3+2)*3+d] = v0enc[n*3+d];
    }
}

// ---------------------------------------------------------------------------
// 2. brute-force top-64 nearest neighbors (one thread per query, lists in LDS)
// ---------------------------------------------------------------------------
__global__ void topk_kernel(const float* __restrict__ q,
                            const float* __restrict__ p,
                            const float* __restrict__ mask,  // may be null
                            int selfskip,
                            int*   __restrict__ oidx,
                            float* __restrict__ od2,
                            int Nq, int Mp) {
    __shared__ float sd[64 * 65];
    __shared__ int   si[64 * 65];
    int lq = threadIdx.x;                   // 0..63
    int n  = blockIdx.x * 64 + lq;
    float* myd = sd + lq * 65;
    int*   myi = si + lq * 65;
    if (n >= Nq) return;

    for (int i = 0; i < 64; ++i) { myd[i] = BIG_F; myi[i] = 0; }
    float qx = q[n*3], qy = q[n*3+1], qz = q[n*3+2];
    float worst = BIG_F; int wpos = 0;
    for (int j = 0; j < Mp; ++j) {
        if (selfskip && j == n) continue;
        if (mask && !(mask[j] > 0.f)) continue;
        float dx = p[j*3]   - qx;
        float dy = p[j*3+1] - qy;
        float dz = p[j*3+2] - qz;
        float d2 = dx*dx + dy*dy + dz*dz;
        if (d2 < worst) {
            myd[wpos] = d2; myi[wpos] = j;
            float w2 = -1.f; int wp2 = 0;
            for (int i = 0; i < 64; ++i)
                if (myd[i] > w2) { w2 = myd[i]; wp2 = i; }
            worst = w2; wpos = wp2;
        }
    }
    for (int i = 0; i < 64; ++i) { oidx[n*64+i] = myi[i]; od2[n*64+i] = myd[i]; }
}

// ---------------------------------------------------------------------------
// 3. per-(query,neighbor) interpolation prep: ball->cube, trilinear corners
// ---------------------------------------------------------------------------
__device__ void ball_to_cube(float x, float y, float z,
                             float& obx, float& oby, float& obz) {
    float sq   = x*x + y*y + z*z;
    float nrm  = sqrtf(fmaxf(sq, EPS_F));
    float xy   = x*x + y*y;
    bool  polar = 1.25f * z * z > xy;
    float s_p  = sqrtf(3.0f * nrm / (nrm + fabsf(z) + EPS_F));
    float s_e  = nrm / sqrtf(fmaxf(xy, EPS_F));
    float cx   = polar ? x * s_p : x * s_e;
    float cy   = polar ? y * s_p : y * s_e;
    float cz   = polar ? sgnf(z) * nrm : 1.5f * z;
    float nxy  = sqrtf(fmaxf(cx*cx + cy*cy, EPS_F));
    bool  xdom = fabsf(cy) <= fabsf(cx);
    float sx   = (fabsf(cx) > EPS_F) ? cx : 1.0f;
    float sy   = (fabsf(cy) > EPS_F) ? cy : 1.0f;
    float bx1  = sgnf(cx) * nxy;
    float by1  = bx1 * (4.0f / PI_F) * atanf(cy / sx);
    float by2  = sgnf(cy) * nxy;
    float bx2  = by2 * (4.0f / PI_F) * atanf(cx / sy);
    float rbx  = xdom ? bx1 : bx2;
    float rby  = xdom ? by1 : by2;
    if (cx*cx + cy*cy < EPS_F) { rbx = 0.f; rby = 0.f; }
    if (sq < EPS_F) { obx = 0.f; oby = 0.f; obz = 0.f; return; }
    obx = rbx; oby = rby; obz = cz;
}

__global__ void prep_kernel(const float* __restrict__ qpos,
                            const float* __restrict__ ppos,
                            const int*   __restrict__ nidx,
                            const float* __restrict__ nd2,
                            float* __restrict__ ow,    // (Nq,64,8)
                            int*   __restrict__ oc,    // (Nq,64,8)
                            int Nq) {
    int t = blockIdx.x * blockDim.x + threadIdx.x;
    if (t >= Nq * 64) return;
    int n = t >> 6;
    int j = nidx[t];
    float d2 = nd2[t];
    float ux = (ppos[j*3]   - qpos[n*3])   / RADIUS;
    float uy = (ppos[j*3+1] - qpos[n*3+1]) / RADIUS;
    float uz = (ppos[j*3+2] - qpos[n*3+2]) / RADIUS;
    float rsq = ux*ux + uy*uy + uz*uz;
    float win = 0.f;
    if (d2 <= R2_F) {
        float v = 1.0f - rsq;
        v = v * v * v;
        win = fminf(fmaxf(v, 0.f), 1.f);
    }
    float bx, by, bz;
    ball_to_cube(ux, uy, uz, bx, by, bz);
    float cxn = fminf(fmaxf((bx*0.5f + 0.5f) * 3.0f, 0.f), 3.0f);
    float cyn = fminf(fmaxf((by*0.5f + 0.5f) * 3.0f, 0.f), 3.0f);
    float czn = fminf(fmaxf((bz*0.5f + 0.5f) * 3.0f, 0.f), 3.0f);
    int ix = min(max((int)floorf(cxn), 0), 2);
    int iy = min(max((int)floorf(cyn), 0), 2);
    int iz = min(max((int)floorf(czn), 0), 2);
    float fx = cxn - (float)ix, fy = cyn - (float)iy, fz = czn - (float)iz;
    #pragma unroll
    for (int e = 0; e < 8; ++e) {
        int oi = (e >> 2) & 1, oj = (e >> 1) & 1, ol = e & 1;
        float w = (oi ? fx : 1.f - fx) * (oj ? fy : 1.f - fy) * (ol ? fz : 1.f - fz) * win;
        int corner = ((ix + oi) * 4 + (iy + oj)) * 4 + (iz + ol);
        ow[t*8 + e] = w;
        oc[t*8 + e] = corner;
    }
}

// ---------------------------------------------------------------------------
// 3b. filter-bank transpose + f16 quantize: KT[o][l*C+c] = (f16)K[l][c][o]
//     (stage-2 A operand becomes two plain 16B global loads, zero cvt VALU)
// ---------------------------------------------------------------------------
__global__ void ktrans_kernel(const float* __restrict__ k, _Float16* __restrict__ kt,
                              int C, int O) {
    int t = blockIdx.x * blockDim.x + threadIdx.x;
    if (t >= 64 * C * O) return;
    int lc = t / O, o = t - lc * O;
    kt[o * (64 * C) + lc] = (_Float16)k[t];
}

// ---------------------------------------------------------------------------
// f16 fragment K-run mapping (CDNA5 ISA, 16-bit A 16x32):
// lanes 0-15 : halves 0-7 -> K 0..7,   halves 8-15 -> K 16..23
// lanes16-31 : halves 0-7 -> K 8..15,  halves 8-15 -> K 24..31
// => per lane: two contiguous runs of 8 starting at kbase and kbase+16.
// Works for LDS or global pointers (generic address space).
// ---------------------------------------------------------------------------
__device__ __forceinline__ v16h frag_runs(const _Float16* row, int kbase) {
    v8h lo = *(const v8h*)(row + kbase);
    v8h hi = *(const v8h*)(row + kbase + 16);
    return __builtin_shufflevector(lo, hi, 0,1,2,3,4,5,6,7,8,9,10,11,12,13,14,15);
}

// ---------------------------------------------------------------------------
// 4. CConv via WMMA. One workgroup (8 wave32s) per query.
//    Stage 1: cellacc = Wcell^T(64x64) x G(64xF)        (F = S*Cp, Cp%32==0)
//    Stage 2: out[s][o] = sum_{l,c} cellacc[l][s,c] * K[l][c][o]
// LDS: stgW(512f)|stgC(512i)|WT[l][k] f16|GT[f][k] f16|CAT[s][l*Cp+c] f16
// ---------------------------------------------------------------------------
__global__ void __launch_bounds__(256)
cconv_wmma(const float*    __restrict__ feats,   // (srcN,S,C)
           const int*      __restrict__ nidx,    // (Nq,64)
           const float*    __restrict__ nw,      // (Nq,64,8)
           const int*      __restrict__ ncell,   // (Nq,64,8)
           const float*    __restrict__ kern,    // (64,C,O)
           const _Float16* __restrict__ kernT,   // (O,64*C) f16, or null
           const float*    __restrict__ bias,    // (O)
           float* __restrict__ outp,             // (Nq,S,O)
           int S, int C, int Cp, int O, int Nq) {
    extern __shared__ char smemraw[];
    const int F = S * Cp;
    float*    stgW = (float*)smemraw;                 // 512 f32
    int*      stgC = (int*)(smemraw + 2048);          // 512 i32
    _Float16* WT   = (_Float16*)(smemraw + 4096);     // 64*64
    _Float16* GT   = WT + 64 * 64;                    // F*64
    _Float16* CAT  = GT + 64 * F;                     // S*64*Cp == 64*F

    const int n    = blockIdx.x;
    const int tid  = threadIdx.x;
    const int lane = tid & 31;
    const int wave = tid >> 5;
    const int NW   = blockDim.x >> 5;

    __builtin_prefetch(kernT ? (const void*)kernT : (const void*)kern, 0, 1);

    // ---- stage the per-query interpolation tables into LDS ----
#if defined(HAVE_TDM)
    if (tid < 32) {   // wave 0 issues the DMA; TENSORcnt is per-wave
        tdm_load_1d(nw + (size_t)n * 512,
                    (unsigned)((char*)stgW - smemraw), 512u);
        tdm_load_1d(ncell + (size_t)n * 512,
                    (unsigned)((char*)stgC - smemraw), 512u);
        __builtin_amdgcn_s_wait_tensorcnt((short)0);
    }
#else
    for (int i = tid; i < 512; i += blockDim.x) {
        stgW[i] = nw[(size_t)n * 512 + i];
        stgC[i] = ncell[(size_t)n * 512 + i];
    }
#endif
    // zero W^T while the DMA flies
    for (int i = tid; i < 64 * 64 / 2; i += blockDim.x) ((float*)WT)[i] = 0.f;
    __syncthreads();

    // ---- scatter Wcell^T (WT[l*64+k], k-major for A fragments) ----
    if (tid < 64) {
        #pragma unroll
        for (int e = 0; e < 8; ++e) {
            int l = stgC[tid * 8 + e];
            WT[l * 64 + tid] = (_Float16)stgW[tid * 8 + e];
        }
    }
    // ---- gather neighbor features, k-major, channel-padded to Cp ----
    for (int i = tid; i < 64 * F; i += blockDim.x) {
        int k = i & 63, f = i >> 6;
        int s = f / Cp, c = f - s * Cp;
        float v = 0.f;
        if (c < C) {
            int j = nidx[n * 64 + k];
            v = feats[(j * S + s) * C + c];
        }
        GT[f * 64 + k] = (_Float16)v;
    }
    __syncthreads();

    // ---- stage 1: cellacc = W^T x G ----
    const int numFT = F >> 4;
    const int row   = lane & 15;
    const int kbase = (lane < 16) ? 0 : 8;
    const int rbase = (lane < 16) ? 0 : 8;
    for (int t = wave; t < 4 * numFT; t += NW) {
        int lt = t / numFT, ft = t - lt * numFT;
        const _Float16* wrow = WT + (lt * 16 + row) * 64;
        const _Float16* grow = GT + (ft * 16 + row) * 64;
        v8f acc = {};
        #pragma unroll
        for (int kk = 0; kk < 64; kk += 32) {
            v16h a = frag_runs(wrow, kk + kbase);   // A[l][k] = W[k][l]
            v16h b = frag_runs(grow, kk + kbase);   // B[k][f]
            acc = __builtin_amdgcn_wmma_f32_16x16x32_f16(
                      false, a, false, b, (short)0, acc, false, false);
        }
        int f = ft * 16 + row;           // column of D = feature index
        int s = f / Cp, c = f - s * Cp;
        #pragma unroll
        for (int r = 0; r < 8; ++r) {
            int l = lt * 16 + rbase + r;
            CAT[(s * 64 + l) * Cp + c] = (_Float16)acc[r];
        }
    }
    __syncthreads();

    // ---- stage 2: out = cellacc (K=(l,c)) x kernel ----
    const int nK    = (64 * Cp) >> 5;
    const int numOT = (O + 15) >> 4;
    const int scl   = min(row, S - 1);         // clamp: cols s>=S are discarded
    const _Float16* brow = CAT + scl * 64 * Cp;
    const bool fast = (C == Cp) && ((O & 15) == 0) && (kernT != nullptr);

    for (int ot = wave; ot < numOT; ot += NW) {
        v8f acc = {};
        int o = ot * 16 + row;
        if (fast) {
            const _Float16* arow = kernT + (size_t)o * (64 * C);
            for (int j = 0; j < nK; ++j) {
                int kg = j * 32 + kbase;
                v16h a = frag_runs(arow, kg);   // A[o][lc] = K[l][c][o] (f16 bank)
                v16h b = frag_runs(brow, kg);   // B[lc][s]
                acc = __builtin_amdgcn_wmma_f32_16x16x32_f16(
                          false, a, false, b, (short)0, acc, false, false);
            }
        } else {
            for (int j = 0; j < nK; ++j) {
                int kg0 = j * 32 + kbase;
                v16h a;
                #pragma unroll
                for (int h = 0; h < 16; ++h) {
                    int kg = kg0 + ((h < 8) ? h : (h + 8));
                    int l = kg / Cp, c = kg - l * Cp;
                    float av = 0.f;
                    if (c < C && o < O) av = kern[(l * C + c) * O + o];
                    a[h] = (_Float16)av;
                }
                v16h b = frag_runs(brow, kg0);
                acc = __builtin_amdgcn_wmma_f32_16x16x32_f16(
                          false, a, false, b, (short)0, acc, false, false);
            }
        }
        int s = row;
        if (s < S) {
            #pragma unroll
            for (int r = 0; r < 8; ++r) {
                int oo = ot * 16 + rbase + r;
                if (oo < O) outp[(n * S + s) * O + oo] = acc[r] + bias[oo];
            }
        }
    }
}

// ---------------------------------------------------------------------------
// 5. small elementwise / dense helpers
// ---------------------------------------------------------------------------
__global__ void dense_kernel(const float* __restrict__ x, const float* __restrict__ W,
                             const float* __restrict__ b, float* __restrict__ y,
                             int rows, int C, int O) {
    int t = blockIdx.x * blockDim.x + threadIdx.x;
    if (t >= rows * O) return;
    int r = t / O, o = t - r * O;
    float acc = b[o];
    for (int c = 0; c < C; ++c) acc += x[r * C + c] * W[c * O + o];
    y[t] = acc;
}

__global__ void relu_kernel(const float* __restrict__ x, float* __restrict__ y, int cnt) {
    int t = blockIdx.x * blockDim.x + threadIdx.x;
    if (t < cnt) y[t] = fmaxf(x[t], 0.f);
}

__global__ void combine_kernel(const float* __restrict__ a, const float* __restrict__ b,
                               const float* __restrict__ c, float* __restrict__ y, int cnt) {
    int t = blockIdx.x * blockDim.x + threadIdx.x;
    if (t < cnt) y[t] = a[t] + b[t] + (c ? c[t] : 0.f);
}

__global__ void assemble0_kernel(const float* __restrict__ co,  // (N,1,32)
                                 const float* __restrict__ cf,  // (N,3,32)
                                 const float* __restrict__ df,  // (N,3,32)
                                 float* __restrict__ out0,      // (N,3,96)
                                 int N) {
    int t = blockIdx.x * blockDim.x + threadIdx.x;
    if (t >= N * 3) return;
    int n = t / 3;
    float* dst = out0 + t * 96;
    for (int o = 0; o < 32; ++o) dst[o]      = co[n * 32 + o];      // broadcast over s
    for (int o = 0; o < 32; ++o) dst[32 + o] = cf[t * 32 + o];
    for (int o = 0; o < 32; ++o) dst[64 + o] = df[t * 32 + o];
}

__global__ void final_kernel(const float* __restrict__ out4,   // (N,3,3)
                             const float* __restrict__ p1,
                             const float* __restrict__ p0,
                             const float* __restrict__ v0enc,
                             float* __restrict__ out, int N) {
    int n = blockIdx.x * blockDim.x + threadIdx.x;
    if (n >= N) return;
    for (int d = 0; d < 3; ++d) {
        float corr = out4[(n * 3 + 0) * 3 + d] * 0.25f * (1.0f / 16.0f);
        float pc = p1[n * 3 + d] + corr;
        out[n * 3 + d]             = pc;                                  // p_c
        out[3 * N + n * 3 + d]     = (pc - p0[n * 3 + d]) / DT_F;         // v_c
        out[6 * N + n * 6 + 0 + d] = out4[(n * 3 + 1) * 3 + d] * 0.25f;   // m_matrix s=1
        out[6 * N + n * 6 + 3 + d] = out4[(n * 3 + 2) * 3 + d] * 0.25f;   // m_matrix s=2
        out[12 * N + n * 3 + d]    = v0enc[n * 3 + d];                    // state_feats
    }
}

// ---------------------------------------------------------------------------
// host launch
// ---------------------------------------------------------------------------
extern "C" void kernel_launch(void* const* d_in, const int* in_sizes, int n_in,
                              void* d_out, int out_size, void* d_ws, size_t ws_size,
                              hipStream_t stream) {
    const float* v0enc  = (const float*)d_in[1];
    const float* p0     = (const float*)d_in[2];
    const float* v0     = (const float*)d_in[3];
    const float* acc    = (const float*)d_in[4];
    const float* other  = (const float*)d_in[5];
    const float* box    = (const float*)d_in[6];
    const float* boxf   = (const float*)d_in[7];
    const float* boxm   = (const float*)d_in[9];
    const float* k0f    = (const float*)d_in[10];
    const float* b0f    = (const float*)d_in[11];
    const float* k0o    = (const float*)d_in[12];
    const float* b0o    = (const float*)d_in[13];
    const float* wdf    = (const float*)d_in[14];
    const float* bdf    = (const float*)d_in[15];
    const float* kc[4]  = {(const float*)d_in[16], (const float*)d_in[20],
                           (const float*)d_in[24], (const float*)d_in[28]};
    const float* bc[4]  = {(const float*)d_in[17], (const float*)d_in[21],
                           (const float*)d_in[25], (const float*)d_in[29]};
    const float* wd[4]  = {(const float*)d_in[18], (const float*)d_in[22],
                           (const float*)d_in[26], (const float*)d_in[30]};
    const float* bd[4]  = {(const float*)d_in[19], (const float*)d_in[23],
                           (const float*)d_in[27], (const float*)d_in[31]};

    const int N = in_sizes[2] / 3;
    const int M = in_sizes[6] / 3;

    // workspace carve-out
    char* wp = (char*)d_ws;
    auto alloc = [&](size_t bytes) -> void* {
        void* p = (void*)wp;
        wp += (bytes + 255) & ~(size_t)255;
        return p;
    };
    float* p1   = (float*)alloc((size_t)N * 3 * 4);
    float* v1   = (float*)alloc((size_t)N * 3 * 4);
    float* ff   = (float*)alloc((size_t)N * 9 * 4);
    int*   idxF = (int*)  alloc((size_t)N * 64 * 4);
    float* d2F  = (float*)alloc((size_t)N * 64 * 4);
    float* wF   = (float*)alloc((size_t)N * 64 * 8 * 4);
    int*   cF   = (int*)  alloc((size_t)N * 64 * 8 * 4);
    int*   idxB = (int*)  alloc((size_t)N * 64 * 4);
    float* d2B  = (float*)alloc((size_t)N * 64 * 4);
    float* wB   = (float*)alloc((size_t)N * 64 * 8 * 4);
    int*   cB   = (int*)  alloc((size_t)N * 64 * 8 * 4);
    float* co   = (float*)alloc((size_t)N * 32 * 4);
    float* cf   = (float*)alloc((size_t)N * 3 * 32 * 4);
    float* dfb  = (float*)alloc((size_t)N * 3 * 32 * 4);
    float* out0 = (float*)alloc((size_t)N * 3 * 96 * 4);
    float* hbuf = (float*)alloc((size_t)N * 3 * 96 * 4);
    float* ocb  = (float*)alloc((size_t)N * 3 * 64 * 4);
    float* odb  = (float*)alloc((size_t)N * 3 * 64 * 4);
    float* out1 = (float*)alloc((size_t)N * 3 * 64 * 4);
    float* out2 = (float*)alloc((size_t)N * 3 * 64 * 4);
    float* out3 = (float*)alloc((size_t)N * 3 * 64 * 4);
    float* out4 = (float*)alloc((size_t)N * 3 * 3 * 4);
    _Float16* ktbuf = (_Float16*)alloc((size_t)64 * 96 * 64 * 2);  // f16 transposed filters

    const int TB = 256;
    auto blk = [&](int cnt) { return (cnt + TB - 1) / TB; };

    // 1. integrate
    integrate_kernel<<<blk(N), TB, 0, stream>>>(p0, v0, acc, other, v0enc, p1, v1, ff, N);

    // 2. neighbor search
    topk_kernel<<<(N + 63) / 64, 64, 0, stream>>>(p1, p1, nullptr, 1, idxF, d2F, N, N);
    topk_kernel<<<(N + 63) / 64, 64, 0, stream>>>(p1, box, boxm, 0, idxB, d2B, N, M);

    // 3. interpolation prep (shared by every fluid conv layer)
    prep_kernel<<<blk(N * 64), TB, 0, stream>>>(p1, p1, idxF, d2F, wF, cF, N);
    prep_kernel<<<blk(N * 64), TB, 0, stream>>>(p1, box, idxB, d2B, wB, cB, N);

    auto smem = [](int S, int Cp) {
        return (size_t)4096 + 8192 + 2 * (size_t)2 * 64 * S * Cp;
    };

    // 4. layer 0: box conv (S=1,C=1), fluid conv (S=3,C=3), dense branch
    cconv_wmma<<<N, TB, smem(1, 32), stream>>>(boxf, idxB, wB, cB, k0o, nullptr, b0o,
                                               co, 1, 1, 32, 32, N);
    cconv_wmma<<<N, TB, smem(3, 32), stream>>>(ff,   idxF, wF, cF, k0f, nullptr, b0f,
                                               cf, 3, 3, 32, 32, N);
    dense_kernel<<<blk(N * 3 * 32), TB, 0, stream>>>(ff, wdf, bdf, dfb, N * 3, 3, 32);
    assemble0_kernel<<<blk(N * 3), TB, 0, stream>>>(co, cf, dfb, out0, N);

    // 5. layers 1..4
    struct { const float* in; float* out; const float* res; int C, O; } L[4] = {
        { out0, out1, nullptr, 96, 64 },
        { out1, out2, out1,    64, 64 },
        { out2, out3, out2,    64, 64 },
        { out3, out4, nullptr, 64, 3  },
    };
    for (int i = 0; i < 4; ++i) {
        int C = L[i].C, O = L[i].O;
        int Cp = (C + 31) & ~31;
        bool fast = (C == Cp) && ((O & 15) == 0);
        relu_kernel<<<blk(N * 3 * C), TB, 0, stream>>>(L[i].in, hbuf, N * 3 * C);
        if (fast)
            ktrans_kernel<<<blk(64 * C * O), TB, 0, stream>>>(kc[i], ktbuf, C, O);
        cconv_wmma<<<N, TB, smem(3, Cp), stream>>>(hbuf, idxF, wF, cF, kc[i],
                                                   fast ? ktbuf : nullptr, bc[i], ocb,
                                                   3, C, Cp, O, N);
        dense_kernel<<<blk(N * 3 * O), TB, 0, stream>>>(hbuf, wd[i], bd[i], odb, N * 3, C, O);
        combine_kernel<<<blk(N * 3 * O), TB, 0, stream>>>(ocb, odb, L[i].res, L[i].out, N * 3 * O);
    }

    // 6. outputs: p_c, v_c, m_matrix, state_feats
    final_kernel<<<blk(N), TB, 0, stream>>>(out4, p1, p0, v0enc, (float*)d_out, N);
}
